// TiLinear_11785390260255
// MI455X (gfx1250) — 8-bit matrix operand order repaired and checked by reading, compiled, not run on hardware
//
#include <hip/hip_runtime.h>
#include <stdint.h>

typedef __attribute__((ext_vector_type(8))) int v8i;

#define M_DIM 8192
#define K_DIM 4096
#define N_DIM 4096
#define BM 128
#define BN 128
#define BK 64
#define KTILES (K_DIM / BK)
#define LDS_STRIDE 80

__device__ __forceinline__ void async_copy_b128(unsigned lds_addr,
                                                const unsigned char* gaddr) {
  asm volatile("global_load_async_to_lds_b128 %0, %1, off"
               :: "v"(lds_addr), "v"(gaddr)
               : "memory");
}
__device__ __forceinline__ void wait_async0() {
  asm volatile("s_wait_asynccnt 0" ::: "memory");
}

__global__ void pack_i32_to_i8(const int* __restrict__ src,
                               unsigned char* __restrict__ dst,
                               int n4, int* __restrict__ gmax_init) {
  int i = blockIdx.x * blockDim.x + threadIdx.x;
  if (gmax_init && i == 0) *gmax_init = 0;
  if (i < n4) {
    int4 v = ((const int4*)src)[i];
    unsigned int p = (unsigned int)(v.x & 0xFF)
                   | ((unsigned int)(v.y & 0xFF) << 8)
                   | ((unsigned int)(v.z & 0xFF) << 16)
                   | ((unsigned int)(v.w & 0xFF) << 24);
    ((unsigned int*)dst)[i] = p;
  }
}

__global__ __launch_bounds__(256)
__attribute__((amdgpu_waves_per_eu(4)))
void int8_gemm_wmma(const unsigned char* __restrict__ A,
                    const unsigned char* __restrict__ W,
                    int* __restrict__ out,
                    int* __restrict__ gmax) {
  __shared__ unsigned char lsA[2][BM * LDS_STRIDE];
  __shared__ unsigned char lsB[2][BN * LDS_STRIDE];
  __shared__ int redbuf[256];

  const int tid  = threadIdx.x;
  const int wave = tid >> 5;
  const int lane = tid & 31;
  const int half = lane >> 4;
  const int l16  = lane & 15;

  const int mBase = blockIdx.y * BM;
  const int nBase = blockIdx.x * BN;
  const int waveM = (wave >> 1) * 32;
  const int waveN = (wave & 1) * 64;

  v8i c[2][4];
#pragma unroll
  for (int mt = 0; mt < 2; ++mt)
#pragma unroll
    for (int nt = 0; nt < 4; ++nt)
#pragma unroll
      for (int r = 0; r < 8; ++r) c[mt][nt][r] = 0;

  int aoff[8], boff[8];
#pragma unroll
  for (int v = 0; v < 8; ++v) {
    aoff[v] = ((v & 1) * 4) + (((v >> 1) & 1) * 16) + ((v >> 2) * 32) + half * 8;
    boff[v] = ((v & 1) * 4) + (((v >> 1) & 1) * 16) + ((v >> 2) * 32) + half * 8;
  }

  const unsigned char* aSrc[2];
  const unsigned char* wSrc[2];
  unsigned ldsAdst[2][2], ldsBdst[2][2];
#pragma unroll
  for (int i = 0; i < 2; ++i) {
    int chunk = tid + i * 256;
    int row = chunk >> 2;
    int col = (chunk & 3) * 16;
    aSrc[i] = A + (size_t)(mBase + row) * K_DIM + col;
    wSrc[i] = W + (size_t)(nBase + row) * K_DIM + col;
    int off = row * LDS_STRIDE + col;
#pragma unroll
    for (int b = 0; b < 2; ++b) {
      ldsAdst[b][i] = (unsigned)(size_t)(lsA[b] + off);
      ldsBdst[b][i] = (unsigned)(size_t)(lsB[b] + off);
    }
  }

#pragma unroll
  for (int i = 0; i < 2; ++i) {
    async_copy_b128(ldsAdst[0][i], aSrc[i]);
    async_copy_b128(ldsBdst[0][i], wSrc[i]);
  }

#pragma unroll 2
  for (int kt = 0; kt < KTILES; ++kt) {
    const int cur = kt & 1;
    const bool hasNext = (kt + 1) < KTILES;

    wait_async0();
    __syncthreads();

    if (hasNext) {
      int k0n = (kt + 1) * BK;
#pragma unroll
      for (int i = 0; i < 2; ++i) {
        async_copy_b128(ldsAdst[1 - cur][i], aSrc[i] + k0n);
        async_copy_b128(ldsBdst[1 - cur][i], wSrc[i] + k0n);
      }
    }

    v8i afrag[2], bfrag[4];
#pragma unroll
    for (int mt = 0; mt < 2; ++mt) {
      int rowb = (waveM + mt * 16 + l16) * LDS_STRIDE;
#pragma unroll
      for (int v = 0; v < 8; ++v)
        afrag[mt][v] = *(const int*)(lsA[cur] + rowb + aoff[v]);
    }
#pragma unroll
    for (int nt = 0; nt < 4; ++nt) {
      int rowb = (waveN + nt * 16 + l16) * LDS_STRIDE;
#pragma unroll
      for (int v = 0; v < 8; ++v)
        bfrag[nt][v] = *(const int*)(lsB[cur] + rowb + boff[v]);
    }

#pragma unroll
    for (int mt = 0; mt < 2; ++mt)
#pragma unroll
      for (int nt = 0; nt < 4; ++nt)
        c[mt][nt] = __builtin_amdgcn_wmma_i32_16x16x64_iu8(
            true, afrag[mt], true, bfrag[nt], c[mt][nt], false, false);
  }

  int lmax = 0;
#pragma unroll
  for (int mt = 0; mt < 2; ++mt)
#pragma unroll
    for (int nt = 0; nt < 4; ++nt)
#pragma unroll
      for (int r = 0; r < 8; ++r) {
        int val = c[mt][nt][r];
        int gm = mBase + waveM + mt * 16 + r + half * 8;
        int gn = nBase + waveN + nt * 16 + l16;
        out[(size_t)gm * N_DIM + gn] = val;
        int a = val < 0 ? -val : val;
        lmax = a > lmax ? a : lmax;
      }

  redbuf[tid] = lmax;
  __syncthreads();
  for (int s = 128; s > 0; s >>= 1) {
    if (tid < s) {
      int o = redbuf[tid + s];
      if (o > redbuf[tid]) redbuf[tid] = o;
    }
    __syncthreads();
  }
  if (tid == 0) atomicMax(gmax, redbuf[0]);
}

__global__ void finalize_quant(float* __restrict__ out,
                               const int* __restrict__ gmax,
                               const int* __restrict__ exp_in,
                               const int* __restrict__ weight_exp) {
  int maxv = *gmax;
  int bw = (maxv == 0) ? 0 : (int)ceilf(log2f((float)maxv));
  int shift = bw - 7;
  int eff = (shift > 1) ? shift : ((shift == 1) ? 2 : 0);

  size_t i = (size_t)blockIdx.x * blockDim.x + threadIdx.x;
  size_t total = (size_t)M_DIM * N_DIM;
  size_t base = i * 4;
  int* iout = (int*)out;

  if (base < total) {
    int4 v = *(const int4*)(iout + base);
    int vv[4] = {v.x, v.y, v.z, v.w};
    float f[4];
#pragma unroll
    for (int j = 0; j < 4; ++j) {
      int x = vv[j];
      int o;
      if (eff > 0) {
        int rt   = x >> eff;
        int prob = x - (rt << eff);
        int hp   = eff >> 1;
        int qp   = prob >> hp;
        int prn  = prob - (qp << hp);
        if (eff & 1) prn <<= 1;
        int sgn  = (x > 0) - (x < 0);
        int dec  = (qp <= prn) ? 0 : sgn;
        o = rt + dec;
        o = o < -127 ? -127 : (o > 127 ? 127 : o);
      } else {
        o = (int)(signed char)(x & 0xFF);
      }
      f[j] = (float)o;
    }
    float4 fo;
    fo.x = f[0]; fo.y = f[1]; fo.z = f[2]; fo.w = f[3];
    *(float4*)(out + base) = fo;
  }
  if (i == 0) {
    int e = *exp_in + *weight_exp + eff;
    out[total] = (float)(signed char)(e & 0xFF);
  }
}

extern "C" void kernel_launch(void* const* d_in, const int* in_sizes, int n_in,
                              void* d_out, int out_size, void* d_ws, size_t ws_size,
                              hipStream_t stream) {
  const int* act_in     = (const int*)d_in[0];
  const int* exp_in     = (const int*)d_in[1];
  const int* weight     = (const int*)d_in[2];
  const int* weight_exp = (const int*)d_in[3];

  float* out = (float*)d_out;
  unsigned char* ws  = (unsigned char*)d_ws;
  unsigned char* Apk = ws;
  unsigned char* Wpk = ws + (size_t)M_DIM * K_DIM;
  int* gmax = (int*)(ws + (size_t)M_DIM * K_DIM + (size_t)N_DIM * K_DIM);

  int nA4 = M_DIM * K_DIM / 4;
  pack_i32_to_i8<<<(nA4 + 255) / 256, 256, 0, stream>>>(act_in, Apk, nA4, gmax);
  int nW4 = N_DIM * K_DIM / 4;
  pack_i32_to_i8<<<(nW4 + 255) / 256, 256, 0, stream>>>(weight, Wpk, nW4, nullptr);

  dim3 grid(N_DIM / BN, M_DIM / BM);
  int8_gemm_wmma<<<grid, 256, 0, stream>>>(Apk, Wpk, (int*)out, gmax);

  size_t total4 = (size_t)M_DIM * N_DIM / 4;
  finalize_quant<<<(total4 + 255) / 256, 256, 0, stream>>>(out, gmax, exp_in, weight_exp);
}
